// RMHSA_58935541236283
// MI455X (gfx1250) — compile-verified, matrix operand order checked by reading, not run on hardware
//
#include <hip/hip_runtime.h>

// ---------------------------------------------------------------------------
// Problem constants (from reference): B=2, S=2048, DM=DA=1024, H=16, DH=64
// ---------------------------------------------------------------------------
#define C_B  2
#define C_S  2048
#define C_DM 1024
#define C_DA 1024
#define C_H  16
#define C_DH 64

typedef __attribute__((ext_vector_type(16))) __bf16 v16bf;
typedef __attribute__((ext_vector_type(8)))  float  v8f;

#define WMMA_BF16(a, b, c) \
  __builtin_amdgcn_wmma_f32_16x16x32_bf16(false, (a), false, (b), (short)0, (c), false, false)

// Tensor Data Mover availability (device pass only; host pass uses fallback).
#if defined(__HIP_DEVICE_COMPILE__) && defined(__has_builtin)
#if __has_builtin(__builtin_amdgcn_tensor_load_to_lds)
#define HAVE_TDM 1
#endif
#endif
#ifndef HAVE_TDM
#define HAVE_TDM 0
#endif

#if HAVE_TDM
typedef __attribute__((ext_vector_type(4))) unsigned int u32x4;
typedef __attribute__((ext_vector_type(8))) int          i32x8;
typedef __attribute__((ext_vector_type(4))) int          i32x4;

// 2D tile DMA: global (row-major, 2-byte elems, row stride stride_d0 elems)
// -> LDS (rows packed contiguously, tile_d0 elems per row).
// Descriptor bitfields per CDNA5 ISA 8.3/8.4 (D# group0 + group1).
static __device__ __forceinline__ void tdm_load_2d(unsigned lds_off, const void* gptr,
                                                   unsigned tile_d0, unsigned tile_d1,
                                                   unsigned tensor_d0, unsigned tensor_d1,
                                                   unsigned stride_d0) {
  const unsigned long long ga = (unsigned long long)(uintptr_t)gptr;
  u32x4 g0;
  g0[0] = 1u;                                                  // count=1, user D#
  g0[1] = lds_off;                                             // lds_addr
  g0[2] = (unsigned)(ga & 0xffffffffull);                      // global_addr lo
  g0[3] = ((unsigned)(ga >> 32) & 0x01ffffffu) | (2u << 30);   // global_addr hi | type=2
  i32x8 g1;
  g1[0] = (int)(1u << 16);                                     // wg_mask=0, data_size=2B
  g1[1] = (int)(tensor_d0 << 16);                              // tensor_dim0 lo16
  g1[2] = (int)((tensor_d0 >> 16) | (tensor_d1 << 16));        // dim0 hi | dim1 lo
  g1[3] = (int)((tensor_d1 >> 16) | (tile_d0 << 16));          // dim1 hi | tile_dim0
  g1[4] = (int)(tile_d1 & 0xffffu);                            // tile_dim1, tile_dim2=0
  g1[5] = (int)stride_d0;                                      // tensor_dim0_stride lo32
  g1[6] = 0;                                                   // stride hi | dim1_stride lo
  g1[7] = 0;
  const i32x4 z4 = {0, 0, 0, 0};
#if defined(__clang_major__) && (__clang_major__ >= 23)
  const i32x8 z8 = {0, 0, 0, 0, 0, 0, 0, 0};
  __builtin_amdgcn_tensor_load_to_lds(g0, g1, z4, z4, z8, 0);
#else
  __builtin_amdgcn_tensor_load_to_lds(g0, g1, z4, z4, 0);
#endif
}
#endif // HAVE_TDM

// 16-element bf16 fragment load: two contiguous 16B chunks (per-lane K-groups
// {0..7,16..23} / {8..15,24..31}; caller applies the +half*8 element bias).
static __device__ __forceinline__ void load_frag16(v16bf& f, const __bf16* p) {
  ((uint4*)&f)[0] = *(const uint4*)(p);
  ((uint4*)&f)[1] = *(const uint4*)(p + 16);
}

// ---------------------------------------------------------------------------
// Elementwise helpers
// ---------------------------------------------------------------------------
__global__ void cvt_f32_to_bf16(const float* __restrict__ src, __bf16* __restrict__ dst, long n) {
  long i = (long)blockIdx.x * blockDim.x + threadIdx.x;
  long stride = (long)gridDim.x * blockDim.x;
  for (; i < n; i += stride) dst[i] = (__bf16)src[i];
}

__global__ void fill_zero_bf16(__bf16* __restrict__ dst, long n) {
  long i = (long)blockIdx.x * blockDim.x + threadIdx.x;
  long stride = (long)gridDim.x * blockDim.x;
  for (; i < n; i += stride) dst[i] = (__bf16)0.0f;
}

// ---------------------------------------------------------------------------
// bf16 GEMM: Y = A(4096x1024) * W(1024x1024) + bias
// mode 0: write bf16 head-major [(b*H+h)*S + s]*DH + d   (for Q/K/V)
// mode 1: write f32 row-major                            (final projection)
// 256 threads = 8 waves; WG tile 128(M) x 64(N); wave = 16(M) x 64(N).
// A tile staged by the Tensor Data Mover; W^T tile staged manually.
// ---------------------------------------------------------------------------
__global__ __launch_bounds__(256)
void gemm_bf16(const __bf16* __restrict__ A, const __bf16* __restrict__ W,
               const float* __restrict__ bias, void* __restrict__ out, int mode) {
  constexpr int Kdim = C_DM, Ndim = C_DA;
  __shared__ __align__(16) __bf16 As[128 * 32];
  __shared__ __align__(16) __bf16 Bs[64 * 32];

  const int t = threadIdx.x;
  const int w = t >> 5, lane = t & 31, half = lane >> 4, lr = lane & 15;
  const int n0 = blockIdx.x * 64;
  const int m0 = blockIdx.y * 128;

  v8f acc[4] = {};

  const int bn = t & 63, bk0 = (t >> 6) * 8;

  for (int k0 = 0; k0 < Kdim; k0 += 32) {
    __syncthreads();
#if HAVE_TDM
    if (w == 0)  // DMA the 128x32 A tile into LDS
      tdm_load_2d((unsigned)(uintptr_t)As, A + (size_t)m0 * Kdim + k0,
                  /*tile*/32, 128, /*tensor*/32, 128, /*stride*/Kdim);
#else
    { // manual A staging (host pass / no-TDM fallback)
      const uint4* src = (const uint4*)(A + (size_t)(m0 + (t >> 1)) * Kdim + k0 + (t & 1) * 16);
      uint4* dst = (uint4*)(As + (t >> 1) * 32 + (t & 1) * 16);
      dst[0] = src[0];
      dst[1] = src[1];
    }
#endif
    { // stage W^T tile (Bs[n][k], 64x32)
#pragma unroll
      for (int j = 0; j < 8; ++j)
        Bs[bn * 32 + bk0 + j] = W[(size_t)(k0 + bk0 + j) * Ndim + n0 + bn];
    }
    if (k0 + 32 < Kdim)
      __builtin_prefetch(W + (size_t)(k0 + 32 + bk0) * Ndim + n0 + bn, 0, 1);
#if HAVE_TDM
    if (w == 0) __builtin_amdgcn_s_wait_tensorcnt(0);
#endif
    __syncthreads();

    v16bf af, bfr[4];
    load_frag16(af, As + (w * 16 + lr) * 32 + half * 8);
#pragma unroll
    for (int nt = 0; nt < 4; ++nt)
      load_frag16(bfr[nt], Bs + (nt * 16 + lr) * 32 + half * 8);
#pragma unroll
    for (int nt = 0; nt < 4; ++nt)
      acc[nt] = WMMA_BF16(af, bfr[nt], acc[nt]);
  }

#pragma unroll
  for (int nt = 0; nt < 4; ++nt) {
    const int col = n0 + nt * 16 + lr;
    const float bval = bias[col];
#pragma unroll
    for (int v = 0; v < 8; ++v) {
      const int row = m0 + w * 16 + v + half * 8; // C layout: M = v (+8 for hi half)
      const float val = acc[nt][v] + bval;
      if (mode == 0) {
        const int h = col >> 6, d = col & 63;
        const int bb = row >> 11, s = row & (C_S - 1);
        ((__bf16*)out)[(((size_t)bb * C_H + h) * C_S + s) * C_DH + d] = (__bf16)val;
      } else {
        ((float*)out)[(size_t)row * C_DM + col] = val;
      }
    }
  }
}

// ---------------------------------------------------------------------------
// Flash-style causal attention with relative position bias.
// Grid: (S/64, B*H); block 128 = 4 waves; wave owns 16 query rows.
// er points at logical Er row 0 inside a buffer zero-padded 16 rows in front.
// K tile staged via Tensor Data Mover; V tile transposed manually into LDS.
// ---------------------------------------------------------------------------
__global__ __launch_bounds__(128)
void attn_rel(const __bf16* __restrict__ qb, const __bf16* __restrict__ kb,
              const __bf16* __restrict__ vb, const __bf16* __restrict__ er,
              __bf16* __restrict__ att_out) {
  constexpr int S = C_S, DH = C_DH;
  __shared__ __align__(16) __bf16 Ks[32 * 64];      // K tile, row-major
  __shared__ __align__(16) __bf16 Vt[64 * 32];      // V tile transposed [n][k]
  __shared__ __align__(16) __bf16 Ps[4][16 * 32];   // per-wave P re-layout buffer

  const int bh = blockIdx.y;
  const int b = bh >> 4, h = bh & 15;
  const int t = threadIdx.x;
  const int w = t >> 5, lane = t & 31, half = lane >> 4, lr = lane & 15;
  const int q0wg = blockIdx.x * 64;
  const int q0 = q0wg + w * 16;

  const __bf16* Q = qb + (size_t)bh * S * DH;
  const __bf16* K = kb + (size_t)bh * S * DH;
  const __bf16* V = vb + (size_t)bh * S * DH;

  // Q fragments (16 rows x 64), split into two K=32 chunks
  v16bf qf0, qf1;
  {
    const __bf16* p = Q + (size_t)(q0 + lr) * DH + half * 8;
    load_frag16(qf0, p);
    load_frag16(qf1, p + 32);
  }

  v8f O[4] = {};
  float m[8], l[8];
#pragma unroll
  for (int v = 0; v < 8; ++v) { m[v] = -1e30f; l[v] = 0.0f; }

  const int ntiles = q0wg / 32 + 2; // keys up to q0wg+63 (causal)
  for (int kt = 0; kt < ntiles; ++kt) {
    const int k0 = kt * 32;
    __syncthreads();
#if HAVE_TDM
    if (w == 0)  // DMA the 32x64 K tile into LDS
      tdm_load_2d((unsigned)(uintptr_t)Ks, K + (size_t)k0 * DH,
                  /*tile*/64, 32, /*tensor*/64, 32, /*stride*/64);
#else
    { // manual K staging (host pass / no-TDM fallback)
      const int row = t >> 2;
      const int cw = (t & 3) * 16;
      const uint4* ksrc = (const uint4*)(K + (size_t)(k0 + row) * DH + cw);
      uint4* kdst = (uint4*)(Ks + row * 64 + cw);
      kdst[0] = ksrc[0];
      kdst[1] = ksrc[1];
    }
#endif
    { // stage transposed V tile (registers -> LDS scatter)
      const int row = t >> 2;
      const int cw = (t & 3) * 16;
      __bf16 vtmp[16];
      const uint4* vsrc = (const uint4*)(V + (size_t)(k0 + row) * DH + cw);
      *(uint4*)&vtmp[0] = vsrc[0];
      *(uint4*)&vtmp[8] = vsrc[1];
#pragma unroll
      for (int i = 0; i < 16; ++i) Vt[(cw + i) * 32 + row] = vtmp[i];
    }
#if HAVE_TDM
    if (w == 0) __builtin_amdgcn_s_wait_tensorcnt(0);
#endif
    __syncthreads();

    // scores S = Q * K^T  (two 16x16 column groups, K=64 -> 2 WMMAs each)
    v8f s0 = {}, s1 = {};
    {
      v16bf kf[4];
      load_frag16(kf[0], Ks + lr * 64 + half * 8);
      load_frag16(kf[1], Ks + lr * 64 + 32 + half * 8);
      load_frag16(kf[2], Ks + (16 + lr) * 64 + half * 8);
      load_frag16(kf[3], Ks + (16 + lr) * 64 + 32 + half * 8);
      s0 = WMMA_BF16(qf0, kf[0], s0);
      s0 = WMMA_BF16(qf1, kf[1], s0);
      s1 = WMMA_BF16(qf0, kf[2], s1);
      s1 = WMMA_BF16(qf1, kf[3], s1);
    }

    // relative bias slabs: P[sf][i][tt] = Q[i] . Er[c-16+16*sf+tt]
    const int c = (S - 1) + k0 - q0;
    v8f P[3] = {};
#pragma unroll
    for (int sf = 0; sf < 3; ++sf) {
      const __bf16* ep = er + (ptrdiff_t)(c - 16 + sf * 16 + lr) * DH + half * 8;
      v16bf ef0, ef1;
      load_frag16(ef0, ep);
      load_frag16(ef1, ep + 32);
      P[sf] = WMMA_BF16(qf0, ef0, P[sf]);
      P[sf] = WMMA_BF16(qf1, ef1, P[sf]);
    }

    // Toeplitz select + causal mask + online softmax (per C-layout row pair)
    float e0s[8], e1s[8];
#pragma unroll
    for (int v = 0; v < 8; ++v) {
      const int i = v + half * 8;
      const int T0 = 16 + lr - i;                 // in [1,31]
      const int sl = ((lr - i) & 15) | (half << 4);
      const float r0 = __shfl(P[0][v], sl, 32);
      const float r1 = __shfl(P[1][v], sl, 32);
      const float r2 = __shfl(P[2][v], sl, 32);
      const float rel0 = (T0 < 16) ? r0 : r1;
      const float rel1 = (T0 < 16) ? r1 : r2;     // T1 = T0 + 16
      const int q = q0 + i;
      const float a0 = (k0 + lr      <= q) ? (s0[v] + rel0) * 0.125f : -1e30f;
      const float a1 = (k0 + lr + 16 <= q) ? (s1[v] + rel1) * 0.125f : -1e30f;

      float mx = fmaxf(a0, a1);
      mx = fmaxf(mx, __shfl_xor(mx, 1, 32));
      mx = fmaxf(mx, __shfl_xor(mx, 2, 32));
      mx = fmaxf(mx, __shfl_xor(mx, 4, 32));
      mx = fmaxf(mx, __shfl_xor(mx, 8, 32));
      const float mnew = fmaxf(m[v], mx);
      const float sc = __expf(m[v] - mnew);
      const float e0 = __expf(a0 - mnew);
      const float e1 = __expf(a1 - mnew);
      float rs = e0 + e1;
      rs += __shfl_xor(rs, 1, 32);
      rs += __shfl_xor(rs, 2, 32);
      rs += __shfl_xor(rs, 4, 32);
      rs += __shfl_xor(rs, 8, 32);
      l[v] = l[v] * sc + rs;
      m[v] = mnew;
      O[0][v] *= sc; O[1][v] *= sc; O[2][v] *= sc; O[3][v] *= sc;
      e0s[v] = e0; e1s[v] = e1;
    }

    // re-layout P (C layout -> A fragment) through per-wave LDS
    __bf16* ps = &Ps[w][0];
#pragma unroll
    for (int v = 0; v < 8; ++v) {
      const int i = v + half * 8;
      ps[i * 32 + lr]      = (__bf16)e0s[v];
      ps[i * 32 + lr + 16] = (__bf16)e1s[v];
    }
    asm volatile("s_wait_dscnt 0" ::: "memory");
    v16bf pf;
    load_frag16(pf, ps + lr * 32 + half * 8);

    // O += P * V  (K=32, four 16-wide output column groups)
#pragma unroll
    for (int f = 0; f < 4; ++f) {
      v16bf vf;
      load_frag16(vf, Vt + (f * 16 + lr) * 32 + half * 8);
      O[f] = WMMA_BF16(pf, vf, O[f]);
    }
  }

  // normalize and store bf16, row-major (b*S+q) x 1024 with cols h*64+n
#pragma unroll
  for (int v = 0; v < 8; ++v) {
    const float inv = 1.0f / l[v];
    const int q = q0 + v + half * 8;
    __bf16* dst = att_out + ((size_t)b * S + q) * C_DA + h * C_DH;
#pragma unroll
    for (int f = 0; f < 4; ++f) dst[f * 16 + lr] = (__bf16)(O[f][v] * inv);
  }
}

// ---------------------------------------------------------------------------
// Host launcher
// ---------------------------------------------------------------------------
extern "C" void kernel_launch(void* const* d_in, const int* in_sizes, int n_in,
                              void* d_out, int out_size, void* d_ws, size_t ws_size,
                              hipStream_t stream) {
  (void)in_sizes; (void)n_in; (void)out_size; (void)ws_size;
  const float* x  = (const float*)d_in[0];
  const float* Wq = (const float*)d_in[1];
  const float* bq = (const float*)d_in[2];
  const float* Wk = (const float*)d_in[3];
  const float* bk = (const float*)d_in[4];
  const float* Wv = (const float*)d_in[5];
  const float* bv = (const float*)d_in[6];
  const float* Wo = (const float*)d_in[7];
  const float* bo = (const float*)d_in[8];
  const float* Er = (const float*)d_in[9];

  const long rows = (long)C_B * C_S;           // 4096
  const long xlen = rows * C_DM;               // 4M
  const long wlen = (long)C_DM * C_DA;         // 1M
  const long erp_rows = 4160;                  // 16 front pad + 4095 + back pad
  const long erlen = (2L * 2048 - 1) * C_DH;   // 4095*64

  char* ws = (char*)d_ws;
  size_t off = 0;
  auto alloc = [&](size_t bytes) -> char* {
    char* p = ws + off;
    off = (off + bytes + 255) & ~(size_t)255;
    return p;
  };
  __bf16* xb  = (__bf16*)alloc((size_t)xlen * 2);
  __bf16* wqb = (__bf16*)alloc((size_t)wlen * 2);
  __bf16* wkb = (__bf16*)alloc((size_t)wlen * 2);
  __bf16* wvb = (__bf16*)alloc((size_t)wlen * 2);
  __bf16* wob = (__bf16*)alloc((size_t)wlen * 2);
  __bf16* erp = (__bf16*)alloc((size_t)erp_rows * C_DH * 2);
  __bf16* qh  = (__bf16*)alloc((size_t)xlen * 2);
  __bf16* kh  = (__bf16*)alloc((size_t)xlen * 2);
  __bf16* vh  = (__bf16*)alloc((size_t)xlen * 2);
  __bf16* ab  = (__bf16*)alloc((size_t)xlen * 2);

  cvt_f32_to_bf16<<<2048, 256, 0, stream>>>(x, xb, xlen);
  cvt_f32_to_bf16<<<1024, 256, 0, stream>>>(Wq, wqb, wlen);
  cvt_f32_to_bf16<<<1024, 256, 0, stream>>>(Wk, wkb, wlen);
  cvt_f32_to_bf16<<<1024, 256, 0, stream>>>(Wv, wvb, wlen);
  cvt_f32_to_bf16<<<1024, 256, 0, stream>>>(Wo, wob, wlen);
  fill_zero_bf16<<<512, 256, 0, stream>>>(erp, erp_rows * C_DH);
  cvt_f32_to_bf16<<<512, 256, 0, stream>>>(Er, erp + 16 * C_DH, erlen);

  dim3 ggrid(C_DA / 64, (unsigned)(rows / 128)); // (16, 32)
  gemm_bf16<<<ggrid, 256, 0, stream>>>(xb, wqb, bq, qh, 0);
  gemm_bf16<<<ggrid, 256, 0, stream>>>(xb, wkb, bk, kh, 0);
  gemm_bf16<<<ggrid, 256, 0, stream>>>(xb, wvb, bv, vh, 0);

  attn_rel<<<dim3(C_S / 64, C_B * C_H), 128, 0, stream>>>(qh, kh, vh, erp + 16 * C_DH, ab);

  gemm_bf16<<<ggrid, 256, 0, stream>>>(ab, wob, bo, d_out, 1);
}
